// SymbolicProjector_71296457113951
// MI455X (gfx1250) — matrix operand order repaired and checked by reading, not compiled, not run on hardware
//
#include <hip/hip_runtime.h>
#include <hip/hip_bf16.h>
#include <stdint.h>

typedef __attribute__((ext_vector_type(16))) __bf16 v16bf;
typedef __attribute__((ext_vector_type(8)))  float  v8f;
typedef __attribute__((ext_vector_type(4)))  float  f32x4;   // native vector: OK for NT builtins

#define DIM             512
#define NCODES          1024
#define TILE            32                    // codes per LDS tile (2 x WMMA N)
#define NTILES          (NCODES / TILE)       // 32
#define KCHUNKS         (DIM / 32)            // 16 WMMA k-steps per tile column
#define WAVES_PER_BLOCK 8
#define THREADS         (WAVES_PER_BLOCK * 32)
#define ROWS_PER_BLOCK  (WAVES_PER_BLOCK * 16)   // 128

// Workspace layout: [0, 1MB) bf16 codebook, [1MB, 1MB+4KB) f32 ||c||^2
#define WS_CB16_BYTES   ((size_t)NCODES * DIM * 2)

// ---- one CDNA5 async global->LDS b128 transfer per lane ----
__device__ __forceinline__ void async_b128(uint32_t lds_byte, uint64_t gaddr) {
  asm volatile("global_load_async_to_lds_b128 %0, %1, off"
               :: "v"(lds_byte), "v"(gaddr) : "memory");
}

// =====================================================================
// Prep: codebook f32 -> bf16 (RNE) + per-code squared norm.
// One wave32 per code; lane handles 16 consecutive dims.
// =====================================================================
__global__ __launch_bounds__(32)
void vq_prep(const float* __restrict__ cb, __bf16* __restrict__ cb16,
             float* __restrict__ c2) {
  const int code = blockIdx.x;
  const int lane = threadIdx.x;
  const float* src = cb + (size_t)code * DIM + lane * 16;
  f32x4 f0 = ((const f32x4*)src)[0];
  f32x4 f1 = ((const f32x4*)src)[1];
  f32x4 f2 = ((const f32x4*)src)[2];
  f32x4 f3 = ((const f32x4*)src)[3];

  float s = f0.x*f0.x + f0.y*f0.y + f0.z*f0.z + f0.w*f0.w
          + f1.x*f1.x + f1.y*f1.y + f1.z*f1.z + f1.w*f1.w
          + f2.x*f2.x + f2.y*f2.y + f2.z*f2.z + f2.w*f2.w
          + f3.x*f3.x + f3.y*f3.y + f3.z*f3.z + f3.w*f3.w;

  alignas(16) __bf16 b[16];
  b[0]=(__bf16)f0.x;  b[1]=(__bf16)f0.y;  b[2]=(__bf16)f0.z;  b[3]=(__bf16)f0.w;
  b[4]=(__bf16)f1.x;  b[5]=(__bf16)f1.y;  b[6]=(__bf16)f1.z;  b[7]=(__bf16)f1.w;
  b[8]=(__bf16)f2.x;  b[9]=(__bf16)f2.y;  b[10]=(__bf16)f2.z; b[11]=(__bf16)f2.w;
  b[12]=(__bf16)f3.x; b[13]=(__bf16)f3.y; b[14]=(__bf16)f3.z; b[15]=(__bf16)f3.w;

  __bf16* dst = cb16 + (size_t)code * DIM + lane * 16;
  ((uint4*)dst)[0] = ((const uint4*)b)[0];
  ((uint4*)dst)[1] = ((const uint4*)b)[1];

#pragma unroll
  for (int m = 16; m >= 1; m >>= 1) s += __shfl_xor(s, m, 32);
  if (lane == 0) c2[code] = s;
}

// =====================================================================
// Main kernel.
// =====================================================================
// Stage one 32-code bf16 tile (32 KB): 256 threads x 8 async b128.
__device__ __forceinline__ void stage_tile(const __bf16* __restrict__ cb16,
                                           const __bf16* lds_dst,
                                           int tile, int tid) {
  uint32_t lds0 = (uint32_t)(uintptr_t)lds_dst;
  uint64_t g0   = (uint64_t)(uintptr_t)(cb16 + (size_t)tile * TILE * DIM);
#pragma unroll
  for (int q = 0; q < 8; ++q) {
    uint32_t off = (uint32_t)(tid + q * THREADS) * 16u;
    async_b128(lds0 + off, g0 + (uint64_t)off);
  }
}

__global__ __launch_bounds__(THREADS, 1)
void vq_argmin_gather(const float* __restrict__ x,
                      const __bf16* __restrict__ cb16,
                      const float* __restrict__ c2g,
                      const float* __restrict__ cb,
                      float* __restrict__ out) {
  __shared__ __align__(64) __bf16 tileB[2][TILE * DIM];   // 2 x 32 KB
  __shared__ __align__(16) float  lds_c2[NCODES];         // 4 KB

  const int tid  = threadIdx.x;
  const int wave = tid >> 5;
  const int lane = tid & 31;
  const int half = lane >> 4;
  const int l16  = lane & 15;

  const size_t rowBase = (size_t)blockIdx.x * ROWS_PER_BLOCK + (size_t)wave * 16;

  // Stage ||c||^2 table (4 KB = 256 x b128) + first tile asynchronously.
  async_b128((uint32_t)(uintptr_t)&lds_c2[0] + (uint32_t)tid * 16u,
             (uint64_t)(uintptr_t)c2g + (uint64_t)tid * 16u);
  stage_tile(cb16, &tileB[0][0], 0, tid);

  // ---- Load this wave's 16x512 x tile into resident A fragments (NT: x read once) ----
  // A k order used here (contiguous, same as B): lanes 0-15 row=l16, K 0..15; lanes 16-31 K 16..31.
  const float* xrow = x + (rowBase + (size_t)l16) * DIM;
  v16bf A[KCHUNKS];
#pragma unroll
  for (int kk = 0; kk < KCHUNKS; ++kk) {
    const int d0 = kk * 32 + half * 16;
    f32x4 f0 = __builtin_nontemporal_load((const f32x4*)(xrow + d0));
    f32x4 f1 = __builtin_nontemporal_load((const f32x4*)(xrow + d0 + 4));
    f32x4 f2 = __builtin_nontemporal_load((const f32x4*)(xrow + d0 + 8));
    f32x4 f3 = __builtin_nontemporal_load((const f32x4*)(xrow + d0 + 12));
    v16bf a;
    a[0]=(__bf16)f0.x;  a[1]=(__bf16)f0.y;  a[2]=(__bf16)f0.z;  a[3]=(__bf16)f0.w;
    a[4]=(__bf16)f1.x;  a[5]=(__bf16)f1.y;  a[6]=(__bf16)f1.z;  a[7]=(__bf16)f1.w;
    a[8]=(__bf16)f2.x;  a[9]=(__bf16)f2.y;  a[10]=(__bf16)f2.z; a[11]=(__bf16)f2.w;
    a[12]=(__bf16)f3.x; a[13]=(__bf16)f3.y; a[14]=(__bf16)f3.z; a[15]=(__bf16)f3.w;
    A[kk] = a;
  }

  float best_d[8];
  int   best_i[8];
#pragma unroll
  for (int r = 0; r < 8; ++r) { best_d[r] = __builtin_inff(); best_i[r] = 0; }

  for (int t = 0; t < NTILES; ++t) {
    // stage(t) fully landed (only its ops outstanding at this point)
    asm volatile("s_wait_asynccnt 0x0" ::: "memory");
    // One barrier: proves (a) every wave's stage(t) done, (b) every wave
    // finished reading buffer (t+1)&1 during compute(t-1) -> safe to restage.
    __syncthreads();
    if (t + 1 < NTILES) stage_tile(cb16, &tileB[(t + 1) & 1][0], t + 1, tid);

    // B k order used here (contiguous, same as A): lane col n=l16; lanes 0-15 K 0..15, lanes 16-31 K 16..31.
    const __bf16* base = &tileB[t & 1][0];
    const __bf16* bt0  = base + (size_t)l16 * DIM + half * 16;          // codes t*32 + 0..15
    const __bf16* bt1  = base + (size_t)(16 + l16) * DIM + half * 16;   // codes t*32 + 16..31
    v8f acc0 = {0.f, 0.f, 0.f, 0.f, 0.f, 0.f, 0.f, 0.f};
    v8f acc1 = {0.f, 0.f, 0.f, 0.f, 0.f, 0.f, 0.f, 0.f};
#pragma unroll
    for (int kk = 0; kk < KCHUNKS; ++kk) {
      v16bf b0 = *(const v16bf*)(bt0 + kk * 32);          // 2x ds_load_b128
      v16bf b1 = *(const v16bf*)(bt1 + kk * 32);
      acc0 = __builtin_amdgcn_wmma_f32_16x16x32_bf16(
                 false, A[kk], false, b0, (short)0, acc0, false, false);
      acc1 = __builtin_amdgcn_wmma_f32_16x16x32_bf16(
                 false, A[kk], false, b1, (short)0, acc1, false, false);
    }

    const int   cand0 = t * TILE + l16;
    const int   cand1 = cand0 + 16;
    const float c2v0  = lds_c2[cand0];
    const float c2v1  = lds_c2[cand1];
#pragma unroll
    for (int r = 0; r < 8; ++r) {
      float d0 = __builtin_fmaf(-2.f, acc0[r], c2v0);     // ||c||^2 - 2 x.c
      if (d0 < best_d[r]) { best_d[r] = d0; best_i[r] = cand0; }
      float d1 = __builtin_fmaf(-2.f, acc1[r], c2v1);
      if (d1 < best_d[r]) { best_d[r] = d1; best_i[r] = cand1; }
    }
  }

  // ---- cross-lane argmin (first-occurrence ties) + gather f32 codebook rows ----
#pragma unroll
  for (int r = 0; r < 8; ++r) {
    float bd = best_d[r]; int bi = best_i[r];
#pragma unroll
    for (int m = 8; m >= 1; m >>= 1) {                    // within each 16-lane half
      float od = __shfl_xor(bd, m, 32);
      int   oi = __shfl_xor(bi, m, 32);
      if (od < bd || (od == bd && oi < bi)) { bd = od; bi = oi; }
    }
    const int idx_lo = __shfl(bi, 0, 32);                 // lanes 0-15 -> row r
    const int idx_hi = __shfl(bi, 16, 32);                // lanes 16-31 -> row r+8
    {
      const f32x4* s  = (const f32x4*)(cb + (size_t)idx_lo * DIM);
      f32x4*       d4 = (f32x4*)(out + (rowBase + r) * DIM);
#pragma unroll
      for (int q = 0; q < 4; ++q)
        __builtin_nontemporal_store(s[lane + q * 32], &d4[lane + q * 32]);
    }
    {
      const f32x4* s  = (const f32x4*)(cb + (size_t)idx_hi * DIM);
      f32x4*       d4 = (f32x4*)(out + (rowBase + r + 8) * DIM);
#pragma unroll
      for (int q = 0; q < 4; ++q)
        __builtin_nontemporal_store(s[lane + q * 32], &d4[lane + q * 32]);
    }
  }
}

extern "C" void kernel_launch(void* const* d_in, const int* in_sizes, int n_in,
                              void* d_out, int out_size, void* d_ws, size_t ws_size,
                              hipStream_t stream) {
  const float* x   = (const float*)d_in[0];   // (8, 8192, 512) f32
  const float* cb  = (const float*)d_in[1];   // (1024, 512) f32
  float*       out = (float*)d_out;           // (8, 8192, 512) f32

  __bf16* cb16 = (__bf16*)d_ws;                                  // 1 MB
  float*  c2   = (float*)((char*)d_ws + WS_CB16_BYTES);          // 4 KB
  (void)ws_size;  // requires ~1.004 MB of scratch

  vq_prep<<<dim3(NCODES), dim3(32), 0, stream>>>(cb, cb16, c2);

  const int nrows = in_sizes[0] / DIM;        // 65536
  dim3 grid(nrows / ROWS_PER_BLOCK);          // 512 blocks of 256 threads (8 wave32)
  vq_argmin_gather<<<grid, THREADS, 0, stream>>>(x, cb16, c2, cb, out);
}
